// att_layer_72550587564439
// MI455X (gfx1250) — matrix operand order repaired and checked by reading, not compiled, not run on hardware
//
#include <hip/hip_runtime.h>
#include <hip/hip_bf16.h>

// ---------------------------------------------------------------------------
// CDNA5 (gfx1250) wave32 WMMA attention:
//   qh = q@Wq+bq ; kh = k@Wk+bk ; vT = (v@Wv+bv)^T   (bf16, in workspace)
//   flash-attention with async global->LDS double-buffered K/V staging
//   out = O @ Wo + bo (fp32)
// ---------------------------------------------------------------------------

typedef __attribute__((ext_vector_type(16))) __bf16 v16bf;
typedef __attribute__((ext_vector_type(8)))  float  v8f;

__device__ __forceinline__ v8f wmma_bf16(v16bf a, v16bf b, v8f c) {
    return __builtin_amdgcn_wmma_f32_16x16x32_bf16(
        /*neg_a=*/false, a, /*neg_b=*/false, b,
        /*c_mod=*/(short)0, c, /*reuse_a=*/false, /*reuse_b=*/false);
}

// 32 contiguous bytes (16 bf16) -> B-fragment lane data (K = half*16 + e)
__device__ __forceinline__ v16bf load_bfrag(const __bf16* p) {
    v16bf r;
    ((uint4*)&r)[0] = *(const uint4*)(p);
    ((uint4*)&r)[1] = *(const uint4*)(p + 8);
    return r;
}

// A-fragment for one 16x32 tile: element e -> K = half*16 + e
// (contiguous order, matching load_bfrag)
__device__ __forceinline__ v16bf load_afrag(const __bf16* row, int half) {
    v16bf r;
    ((uint4*)&r)[0] = *(const uint4*)(row + half * 16);
    ((uint4*)&r)[1] = *(const uint4*)(row + half * 16 + 8);
    return r;
}

// Async 16B global -> LDS copy (per lane). ASYNCcnt-tracked DMA-style copy;
// inline asm keeps it portable across ROCm7.2 / amdgpu-toolchain builtins.
__device__ __forceinline__ void async_copy16(const __bf16* lds_dst, const void* gsrc) {
    unsigned int lds = (unsigned int)(unsigned long long)lds_dst;
    asm volatile("global_load_async_to_lds_b128 %0, %1, off"
                 :: "v"(lds), "v"(gsrc) : "memory");
}
__device__ __forceinline__ void wait_asynccnt_le8() {
    asm volatile("s_wait_asynccnt 0x8" ::: "memory");
}

// ---------------------------------------------------------------------------
// Generic tiled GEMM:  out[nrows x NCOL] = A[nrows x KDIM] * W[KDIM x NCOL] + b
// Block = 256 threads (8 waves), 128 rows per block.
// OUTMODE: 0 = fp32 row-major, 1 = bf16 row-major, 2 = bf16 transposed.
// ---------------------------------------------------------------------------
template <bool ABF16, int KDIM, int NCOL, int OUTMODE>
__global__ __launch_bounds__(256) void gemm_bias_kernel(
    const void* __restrict__ Av, const float* __restrict__ W,
    const float* __restrict__ bias, void* __restrict__ out, int nrows)
{
    constexpr int LDA = 40;                      // bf16 stride: 80B, 16B aligned
    __shared__ alignas(16) __bf16 As[128 * LDA];
    __shared__ alignas(16) __bf16 Bs[NCOL * LDA];

    const int tid   = threadIdx.x;
    const int lane  = tid & 31;
    const int wave  = tid >> 5;
    const int l16   = lane & 15;
    const int lhalf = lane >> 4;
    const int row0  = blockIdx.x * 128;

    const v8f vzero = {0.f, 0.f, 0.f, 0.f, 0.f, 0.f, 0.f, 0.f};
    v8f acc[NCOL / 16];
#pragma unroll
    for (int j = 0; j < NCOL / 16; ++j) acc[j] = vzero;

    for (int kc = 0; kc < KDIM / 32; ++kc) {
        // Stage A tile (128 x 32) -> bf16.
#pragma unroll
        for (int i = 0; i < 16; ++i) {
            int e = tid + 256 * i;               // 0..4095
            int r = e >> 5, c = e & 31;
            size_t gi = (size_t)(row0 + r) * KDIM + kc * 32 + c;
            float a = ABF16 ? (float)((const __bf16*)Av)[gi]
                            : ((const float*)Av)[gi];
            As[r * LDA + c] = (__bf16)a;
        }
        // Stage W tile transposed: Bs[c][r] = W[kc*32+r][c]
#pragma unroll
        for (int i = 0; i < NCOL / 8; ++i) {
            int e = tid + 256 * i;
            int r = e / NCOL, c = e % NCOL;
            Bs[c * LDA + r] = (__bf16)W[(size_t)(kc * 32 + r) * NCOL + c];
        }
        __syncthreads();

        v16bf a = load_afrag(&As[(wave * 16 + l16) * LDA], lhalf);
#pragma unroll
        for (int j = 0; j < NCOL / 16; ++j) {
            v16bf b = load_bfrag(&Bs[(j * 16 + l16) * LDA + lhalf * 16]);
            acc[j] = wmma_bf16(a, b, acc[j]);
        }
        __syncthreads();
    }

#pragma unroll
    for (int j = 0; j < NCOL / 16; ++j) {
        int col = j * 16 + l16;
        float bv = bias[col];
#pragma unroll
        for (int r = 0; r < 8; ++r) {
            int row = row0 + wave * 16 + r + 8 * lhalf;
            float val = acc[j][r] + bv;
            if (OUTMODE == 0)
                ((float*)out)[(size_t)row * NCOL + col] = val;
            else if (OUTMODE == 1)
                ((__bf16*)out)[(size_t)row * NCOL + col] = (__bf16)val;
            else
                ((__bf16*)out)[(size_t)col * nrows + row] = (__bf16)val;
        }
    }
}

// ---------------------------------------------------------------------------
// Flash attention, KV block = 64, K/V tiles async-staged into LDS with
// double buffering. 8 waves/block; each wave owns 16 query rows -> softmax
// state is wave-private (reduced across 16-lane halves with shfl_xor).
// LDS strides chosen bank-conflict-free & 16B aligned:
//   K rows: 136 bf16 (272B), V/P rows: 72 bf16 (144B).
// ---------------------------------------------------------------------------
__global__ __launch_bounds__(256) void flash_attn_kernel(
    const __bf16* __restrict__ qh, const __bf16* __restrict__ kh,
    const __bf16* __restrict__ vT, __bf16* __restrict__ ao,
    int n, float scale)
{
    constexpr int H = 128;
    constexpr int LDK = 136;   // K tile row stride (bf16 elems)
    constexpr int LDV = 72;    // V tile row stride
    constexpr int LDP = 72;    // P tile row stride
    __shared__ alignas(16) __bf16 Kbuf[2][64 * LDK];
    __shared__ alignas(16) __bf16 Vbuf[2][128 * LDV];
    __shared__ alignas(16) __bf16 Pbuf[8][16 * LDP];

    const int tid   = threadIdx.x;
    const int lane  = tid & 31;
    const int wave  = tid >> 5;
    const int l16   = lane & 15;
    const int lhalf = lane >> 4;
    const int q0    = blockIdx.x * 128 + wave * 16;

    // Q fragments (16x128 strip) stay in registers for the whole kernel.
    v16bf qf[4];
#pragma unroll
    for (int kc = 0; kc < 4; ++kc)
        qf[kc] = load_afrag(qh + (size_t)(q0 + l16) * H + kc * 32, lhalf);

    const v8f vzero = {0.f, 0.f, 0.f, 0.f, 0.f, 0.f, 0.f, 0.f};
    v8f O[8];
#pragma unroll
    for (int j = 0; j < 8; ++j) O[j] = vzero;
    float mrow[8], lrow[8];
#pragma unroll
    for (int r = 0; r < 8; ++r) { mrow[r] = -3.0e38f; lrow[r] = 0.f; }

    __bf16* pw = Pbuf[wave];
    const int nIter = n >> 6;            // KV blocks of 64

    // ---- async prefetch of one KV block (K: 16KB contiguous, V: 128x128B) ----
    auto prefetch = [&](int buf, int kv0) {
#pragma unroll
        for (int i = 0; i < 4; ++i) {    // K tile: 1024 x 16B chunks
            int c = wave * 128 + i * 32 + lane;
            int row = c >> 4, piece = c & 15;
            const char* g = (const char*)kh + (size_t)kv0 * 256 + (size_t)c * 16;
            async_copy16(&Kbuf[buf][row * LDK + piece * 8], g);
        }
#pragma unroll
        for (int i = 0; i < 4; ++i) {    // V tile: 128 rows x 8 chunks
            int c = wave * 128 + i * 32 + lane;
            int row = c >> 3, piece = c & 7;
            const char* g = (const char*)vT + (size_t)row * (size_t)n * 2
                          + (size_t)kv0 * 2 + (size_t)piece * 16;
            async_copy16(&Vbuf[buf][row * LDV + piece * 8], g);
        }
    };

    prefetch(0, 0);

    for (int it = 0; it < nIter; ++it) {
        const int buf = it & 1;
        const int kvnext = ((it + 1) << 6) & (n - 1);   // wraps; keeps wait uniform
        prefetch(buf ^ 1, kvnext);
        wait_asynccnt_le8();             // previous 8 (current buf) complete
        __syncthreads();

        const __bf16* Kb = Kbuf[buf];
        const __bf16* Vb = Vbuf[buf];

        // ---- S[16x64] = Q (16x128) @ K_blk^T : 16 WMMAs ----
        v8f s[4];
#pragma unroll
        for (int cc = 0; cc < 4; ++cc) s[cc] = vzero;
#pragma unroll
        for (int cc = 0; cc < 4; ++cc)
#pragma unroll
            for (int kc = 0; kc < 4; ++kc) {
                v16bf kf = load_bfrag(&Kb[(cc * 16 + l16) * LDK + kc * 32 + lhalf * 16]);
                s[cc] = wmma_bf16(qf[kc], kf, s[cc]);
            }

        // ---- online softmax over 64 columns ----
        float alpha[8];
#pragma unroll
        for (int r = 0; r < 8; ++r) {
            float a0 = s[0][r] * scale, a1 = s[1][r] * scale;
            float a2 = s[2][r] * scale, a3 = s[3][r] * scale;
            float rm = fmaxf(fmaxf(a0, a1), fmaxf(a2, a3));
            rm = fmaxf(rm, __shfl_xor(rm, 1, 32));
            rm = fmaxf(rm, __shfl_xor(rm, 2, 32));
            rm = fmaxf(rm, __shfl_xor(rm, 4, 32));
            rm = fmaxf(rm, __shfl_xor(rm, 8, 32));
            float mnew = fmaxf(mrow[r], rm);
            float al = __expf(mrow[r] - mnew);
            float p0 = __expf(a0 - mnew), p1 = __expf(a1 - mnew);
            float p2 = __expf(a2 - mnew), p3 = __expf(a3 - mnew);
            float ps = (p0 + p1) + (p2 + p3);
            ps += __shfl_xor(ps, 1, 32);
            ps += __shfl_xor(ps, 2, 32);
            ps += __shfl_xor(ps, 4, 32);
            ps += __shfl_xor(ps, 8, 32);
            lrow[r] = lrow[r] * al + ps;
            mrow[r] = mnew;
            alpha[r] = al;
            // C layout -> LDS row-major P tile (16 x 64)
            int prow = (r + 8 * lhalf) * LDP;
            pw[prow + l16]      = (__bf16)p0;
            pw[prow + 16 + l16] = (__bf16)p1;
            pw[prow + 32 + l16] = (__bf16)p2;
            pw[prow + 48 + l16] = (__bf16)p3;
        }

        // rescale accumulators
#pragma unroll
        for (int j = 0; j < 8; ++j) {
            v8f t = O[j];
#pragma unroll
            for (int r = 0; r < 8; ++r) t[r] *= alpha[r];
            O[j] = t;
        }

        // ---- O += P (16x64) @ V_blk (64x128) : 16 WMMAs ----
#pragma unroll
        for (int kk = 0; kk < 2; ++kk) {
            v16bf pf = load_afrag(pw + l16 * LDP + kk * 32, lhalf);
#pragma unroll
            for (int j = 0; j < 8; ++j) {
                v16bf vf = load_bfrag(&Vb[(j * 16 + l16) * LDV + kk * 32 + lhalf * 16]);
                O[j] = wmma_bf16(pf, vf, O[j]);
            }
        }
        __syncthreads();                 // everyone done reading buf
    }

    // ---- normalize + store bf16 attention output ----
#pragma unroll
    for (int j = 0; j < 8; ++j) {
        int col = j * 16 + l16;
#pragma unroll
        for (int r = 0; r < 8; ++r) {
            int row = q0 + r + 8 * lhalf;
            ao[(size_t)row * H + col] = (__bf16)(O[j][r] / lrow[r]);
        }
    }
}

// ---------------------------------------------------------------------------
extern "C" void kernel_launch(void* const* d_in, const int* in_sizes, int n_in,
                              void* d_out, int out_size, void* d_ws, size_t ws_size,
                              hipStream_t stream) {
    const float* q  = (const float*)d_in[0];
    const float* k  = (const float*)d_in[1];
    const float* v  = (const float*)d_in[2];
    const float* Wq = (const float*)d_in[3];
    const float* bq = (const float*)d_in[4];
    const float* Wk = (const float*)d_in[5];
    const float* bk = (const float*)d_in[6];
    const float* Wv = (const float*)d_in[7];
    const float* bv = (const float*)d_in[8];
    const float* Wo = (const float*)d_in[9];
    const float* bo = (const float*)d_in[10];

    const int N = in_sizes[0] / 256;          // 16384
    const int nb = N / 128;

    __bf16* qh  = (__bf16*)d_ws;              // [N,128] bf16
    __bf16* khs = qh + (size_t)N * 128;       // [N,128] bf16
    __bf16* vTт = khs + (size_t)N * 128;      // [128,N] bf16 (transposed)
    __bf16* ao  = vTт + (size_t)N * 128;      // [N,128] bf16

    gemm_bias_kernel<false, 256, 128, 1><<<nb, 256, 0, stream>>>(q, Wq, bq, qh, N);
    gemm_bias_kernel<false, 256, 128, 1><<<nb, 256, 0, stream>>>(k, Wk, bk, khs, N);
    gemm_bias_kernel<false, 256, 128, 2><<<nb, 256, 0, stream>>>(v, Wv, bv, vTт, N);

    flash_attn_kernel<<<nb, 256, 0, stream>>>(qh, khs, vTт, ao, N,
                                              0.08838834764831845f /* 128^-0.5 */);

    gemm_bias_kernel<true, 128, 256, 0><<<nb, 256, 0, stream>>>(ao, Wo, bo, d_out, N);
}
